// N2JNet_7662221656189
// MI455X (gfx1250) — matrix operand order repaired and checked by reading, not compiled, not run on hardware
//
#include <hip/hip_runtime.h>

typedef __attribute__((ext_vector_type(16))) _Float16 v16h;
typedef __attribute__((ext_vector_type(8)))  _Float16 v8h;
typedef __attribute__((ext_vector_type(8)))  float    v8f;

#define DL   16
#define DG   16
#define DH   20
#define DPA  20
#define LSTR 40   // f16 row stride in LDS staging buffers (80B, 16B-aligned, conflict-free)

// ---------------------------------------------------------------------------
// WMMA helpers (wave32, one 16-node tile per wave)
// ---------------------------------------------------------------------------
__device__ __forceinline__ v8f wmma_f16(v16h a, v16h b, v8f c) {
  return __builtin_amdgcn_wmma_f32_16x16x32_f16(false, a, false, b, (short)0, c,
                                                false, false);
}

// A fragment (16x32 f16): lane<16 holds row m=lane, K 0..15;
// lane>=16 holds row m=lane-16, K 16..31.
__device__ __forceinline__ v16h load_a_frag(const _Float16* buf, int lane) {
  const int m  = lane & 15;
  const int hi = lane >> 4;
  const _Float16* row = buf + m * LSTR;
  v8h lo8 = *(const v8h*)(row + (hi ? 16 : 0));
  v8h hi8 = *(const v8h*)(row + (hi ? 24 : 8));
  return __builtin_shufflevector(lo8, hi8, 0, 1, 2, 3, 4, 5, 6, 7,
                                 8, 9, 10, 11, 12, 13, 14, 15);
}

// B fragment from pre-packed tile: lane's 16 halfs contiguous (2x b128 loads).
__device__ __forceinline__ v16h load_b_frag(const _Float16* tile, int lane) {
  const _Float16* p = tile + lane * 16;
  v8h lo8 = *(const v8h*)(p);
  v8h hi8 = *(const v8h*)(p + 8);
  return __builtin_shufflevector(lo8, hi8, 0, 1, 2, 3, 4, 5, 6, 7,
                                 8, 9, 10, 11, 12, 13, 14, 15);
}

// One MLP layer with 2 N-tiles (nvalid<=32 outputs), bias+relu, result -> f16 LDS
// with columns >= nvalid zeroed (so it can feed the next WMMA's K padding).
__device__ __forceinline__ void layer2t(const _Float16* in, _Float16* out,
                                        const _Float16* w, const float* bias,
                                        int nvalid, int lane) {
  v16h a  = load_a_frag(in, lane);
  v16h b0 = load_b_frag(w, lane);
  v16h b1 = load_b_frag(w + 512, lane);
  v8f z = {};
  v8f c0 = wmma_f16(a, b0, z);
  v8f c1 = wmma_f16(a, b1, z);
  const int n  = lane & 15;
  const int m0 = (lane >> 4) << 3;
  const float bb0 = bias[n];
  const bool  ok1 = (16 + n) < nvalid;
  const float bb1 = ok1 ? bias[16 + n] : 0.f;
#pragma unroll
  for (int r = 0; r < 8; ++r) {
    float v0 = fmaxf(c0[r] + bb0, 0.f);
    out[(m0 + r) * LSTR + n] = (_Float16)v0;
    float v1 = ok1 ? fmaxf(c1[r] + bb1, 0.f) : 0.f;
    out[(m0 + r) * LSTR + 16 + n] = (_Float16)v1;
  }
}

// Final MLP layer (ntiles N-tiles), bias only (no relu), result -> f32 LDS [16][DPA]
__device__ __forceinline__ void layer_final(const _Float16* in, float* outC,
                                            const _Float16* w, const float* bias,
                                            int ntiles, int nvalid, int lane) {
  v16h a = load_a_frag(in, lane);
  const int n  = lane & 15;
  const int m0 = (lane >> 4) << 3;
  for (int t = 0; t < ntiles; ++t) {
    v16h b = load_b_frag(w + t * 512, lane);
    v8f z = {};
    v8f c = wmma_f16(a, b, z);
    const int nn = t * 16 + n;
    if (nn < nvalid) {
      const float bb = bias[nn];
#pragma unroll
      for (int r = 0; r < 8; ++r) outC[(m0 + r) * DPA + nn] = c[r] + bb;
    }
  }
}

// ---------------------------------------------------------------------------
// Weight packing: fp32 row-major [Ksrc,Nsrc] -> f16 B-fragment tiles (512 halfs)
// dst[tile][lane][j] = W[k][n], n = tile*16 + (lane&15), k = (lane>>4)*16 + j
// ---------------------------------------------------------------------------
__global__ void pack_kernel(const float* __restrict__ W, _Float16* __restrict__ dst,
                            int Ksrc, int Nsrc, int nTiles, int nLayers,
                            long wStride, long dStride) {
  long t = (long)blockIdx.x * blockDim.x + threadIdx.x;
  long total = (long)nLayers * nTiles * 512;
  if (t >= total) return;
  int e = (int)(t & 511);
  long rem = t >> 9;
  int tile  = (int)(rem % nTiles);
  int layer = (int)(rem / nTiles);
  int lane = e >> 4, j = e & 15;
  int n = tile * 16 + (lane & 15);
  int k = ((lane >> 4) << 4) + j;
  float v = 0.f;
  if (k < Ksrc && n < Nsrc) v = W[(long)layer * wStride + (long)k * Nsrc + n];
  dst[(long)layer * dStride + (long)tile * 512 + e] = (_Float16)v;
}

__global__ void zero2_kernel(float* __restrict__ a, long na,
                             float* __restrict__ b, long nb) {
  long t = (long)blockIdx.x * blockDim.x + threadIdx.x;
  if (t < na) a[t] = 0.f;
  if (t < nb) b[t] = 0.f;
}

// ---------------------------------------------------------------------------
// Init MLP: x0[N,8] -> MLP(8,20,20,16) -> LN -> x[N,16]
// ---------------------------------------------------------------------------
__global__ __launch_bounds__(256) void init_kernel(
    const float* __restrict__ x0, float* __restrict__ x,
    const _Float16* __restrict__ pk,
    const float* b1, const float* b2, const float* b3,
    const float* g, const float* be) {
  __shared__ __align__(16) _Float16 sA[8][16 * LSTR];
  __shared__ __align__(16) _Float16 sB[8][16 * LSTR];
  __shared__ __align__(16) float    sC[8][16 * DPA];
  const int wave = threadIdx.x >> 5, lane = threadIdx.x & 31;
  _Float16* bufA = sA[wave];
  _Float16* bufB = sB[wave];
  float*    bufC = sC[wave];
  const int m = lane & 15;
  const long node = ((long)blockIdx.x * 8 + wave) * 16 + m;

  if (lane < 16) {
    const float* xp = x0 + node * 8;
#pragma unroll
    for (int i = 0; i < 8; ++i) bufA[m * LSTR + i] = (_Float16)xp[i];
#pragma unroll
    for (int i = 8; i < 16; ++i) bufA[m * LSTR + i] = (_Float16)0.f;
  } else {
#pragma unroll
    for (int i = 0; i < 16; ++i) bufA[m * LSTR + 16 + i] = (_Float16)0.f;
  }
  __syncthreads();
  layer2t(bufA, bufB, pk + 0, b1, DH, lane);
  __syncthreads();
  layer2t(bufB, bufA, pk + 1024, b2, DH, lane);
  __syncthreads();
  layer_final(bufA, bufC, pk + 2048, b3, 1, DL, lane);
  __syncthreads();
  if (lane < 16) {
    float h[16], mean = 0.f;
#pragma unroll
    for (int i = 0; i < 16; ++i) { h[i] = bufC[m * DPA + i]; mean += h[i]; }
    mean *= (1.f / 16.f);
    float var = 0.f;
#pragma unroll
    for (int i = 0; i < 16; ++i) { float d = h[i] - mean; var += d * d; }
    var *= (1.f / 16.f);
    float rs = rsqrtf(var + 1e-5f);
    float* xp = x + node * DL;
#pragma unroll
    for (int i = 0; i < 16; ++i) xp[i] = (h[i] - mean) * rs * g[i] + be[i];
  }
}

// ---------------------------------------------------------------------------
// Per-iteration node kernel: node model (MLP+LN+residual, in-place x) then
// pre model (MLP+LN) + segmented atomic scatter-add into agg[B,20].
// pk tile offsets: nW1@0 nW2@1024 nW3@2048 pW1@2560 pW2@3584 pW3@4608
// ---------------------------------------------------------------------------
__global__ __launch_bounds__(256) void node_kernel(
    float* __restrict__ x, const int* __restrict__ batch,
    const float* __restrict__ u, float* __restrict__ agg,
    const _Float16* __restrict__ pk,
    const float* nb1, const float* nb2, const float* nb3,
    const float* ng, const float* nbe,
    const float* pb1, const float* pb2, const float* pb3,
    const float* pg, const float* pbe) {
  __shared__ __align__(16) _Float16 sA[8][16 * LSTR];
  __shared__ __align__(16) _Float16 sB[8][16 * LSTR];
  __shared__ __align__(16) float    sC[8][16 * DPA];
  __shared__ int sBid[8][16];
  const int wave = threadIdx.x >> 5, lane = threadIdx.x & 31;
  _Float16* bufA = sA[wave];
  _Float16* bufB = sB[wave];
  float*    bufC = sC[wave];
  const int m = lane & 15;
  const long node = ((long)blockIdx.x * 8 + wave) * 16 + m;

  float xr[16], ur[16];
  if (lane < 16) {
    const float* xp = x + node * DL;
#pragma unroll
    for (int i = 0; i < 16; ++i) { xr[i] = xp[i]; bufA[m * LSTR + i] = (_Float16)xr[i]; }
    sBid[wave][m] = batch[node];
  } else {
    const long b = batch[node];
    const float* up = u + b * DG;
#pragma unroll
    for (int i = 0; i < 16; ++i) { ur[i] = up[i]; bufA[m * LSTR + 16 + i] = (_Float16)ur[i]; }
  }
  __syncthreads();

  // ---- node model: cat(x,u) -> 20 -> 20 -> 16
  layer2t(bufA, bufB, pk + 0, nb1, DH, lane);
  __syncthreads();
  layer2t(bufB, bufA, pk + 1024, nb2, DH, lane);
  __syncthreads();
  layer_final(bufA, bufC, pk + 2048, nb3, 1, DL, lane);
  __syncthreads();

  // ---- LN + residual, write x back, rebuild cat2 = [x_new, u]
  if (lane < 16) {
    float h[16], mean = 0.f;
#pragma unroll
    for (int i = 0; i < 16; ++i) { h[i] = bufC[m * DPA + i]; mean += h[i]; }
    mean *= (1.f / 16.f);
    float var = 0.f;
#pragma unroll
    for (int i = 0; i < 16; ++i) { float d = h[i] - mean; var += d * d; }
    var *= (1.f / 16.f);
    float rs = rsqrtf(var + 1e-5f);
    float* xp = x + node * DL;
#pragma unroll
    for (int i = 0; i < 16; ++i) {
      float xn = (h[i] - mean) * rs * ng[i] + nbe[i] + xr[i];
      xp[i] = xn;
      bufA[m * LSTR + i] = (_Float16)xn;
    }
  } else {
#pragma unroll
    for (int i = 0; i < 16; ++i) bufA[m * LSTR + 16 + i] = (_Float16)ur[i];
  }
  __syncthreads();

  // ---- pre model: cat2 -> 20 -> 20 -> 20
  layer2t(bufA, bufB, pk + 2560, pb1, DH, lane);
  __syncthreads();
  layer2t(bufB, bufA, pk + 3584, pb2, DH, lane);
  __syncthreads();
  layer_final(bufA, bufC, pk + 4608, pb3, 2, DPA, lane);
  __syncthreads();

  // ---- LN on pre (per node), in place in bufC
  if (lane < 16) {
    float h[20], mean = 0.f;
#pragma unroll
    for (int i = 0; i < 20; ++i) { h[i] = bufC[m * DPA + i]; mean += h[i]; }
    mean *= (1.f / 20.f);
    float var = 0.f;
#pragma unroll
    for (int i = 0; i < 20; ++i) { float d = h[i] - mean; var += d * d; }
    var *= (1.f / 20.f);
    float rs = rsqrtf(var + 1e-5f);
#pragma unroll
    for (int i = 0; i < 20; ++i) bufC[m * DPA + i] = (h[i] - mean) * rs * pg[i] + pbe[i];
  }
  __syncthreads();

  // ---- segmented column reduction (batch is sorted) + atomic scatter-add
  if (lane < 20) {
    float acc = bufC[0 * DPA + lane];
    int cur = sBid[wave][0];
#pragma unroll
    for (int r = 1; r < 16; ++r) {
      int bb = sBid[wave][r];
      if (bb != cur) {
        atomicAdd(&agg[(long)cur * DPA + lane], acc);
        acc = 0.f;
        cur = bb;
      }
      acc += bufC[r * DPA + lane];
    }
    atomicAdd(&agg[(long)cur * DPA + lane], acc);
  }
}

// ---------------------------------------------------------------------------
// Per-iteration global model: u = LN(MLP([agg,u])) + u ; agg re-zeroed
// ---------------------------------------------------------------------------
__global__ void global_kernel(float* __restrict__ u, float* __restrict__ agg,
                              const float* qW1, const float* qb1,
                              const float* qW2, const float* qb2,
                              const float* qW3, const float* qb3,
                              const float* qg, const float* qbe, int B) {
  int g = blockIdx.x * blockDim.x + threadIdx.x;
  if (g >= B) return;
  float cat[36];
  for (int i = 0; i < 20; ++i) cat[i] = agg[(long)g * 20 + i];
  for (int i = 0; i < 16; ++i) cat[20 + i] = u[(long)g * 16 + i];
  float h1[20];
  for (int j = 0; j < 20; ++j) {
    float s = qb1[j];
    for (int k = 0; k < 36; ++k) s = fmaf(cat[k], qW1[k * 20 + j], s);
    h1[j] = fmaxf(s, 0.f);
  }
  float h2[20];
  for (int j = 0; j < 20; ++j) {
    float s = qb2[j];
    for (int k = 0; k < 20; ++k) s = fmaf(h1[k], qW2[k * 20 + j], s);
    h2[j] = fmaxf(s, 0.f);
  }
  float h3[16];
  float mean = 0.f;
  for (int j = 0; j < 16; ++j) {
    float s = qb3[j];
    for (int k = 0; k < 20; ++k) s = fmaf(h2[k], qW3[k * 16 + j], s);
    h3[j] = s;
    mean += s;
  }
  mean *= (1.f / 16.f);
  float var = 0.f;
  for (int j = 0; j < 16; ++j) { float d = h3[j] - mean; var += d * d; }
  var *= (1.f / 16.f);
  float rs = rsqrtf(var + 1e-5f);
  for (int j = 0; j < 16; ++j)
    u[(long)g * 16 + j] = (h3[j] - mean) * rs * qg[j] + qbe[j] + cat[20 + j];
  for (int i = 0; i < 20; ++i) agg[(long)g * 20 + i] = 0.f;
}

// ---------------------------------------------------------------------------
// out_local: x[N,16] -> MLP(16,20,20,2) -> out[N,2]
// ---------------------------------------------------------------------------
__global__ __launch_bounds__(256) void outlocal_kernel(
    const float* __restrict__ x, float* __restrict__ out,
    const _Float16* __restrict__ pk,
    const float* b1, const float* b2, const float* b3) {
  __shared__ __align__(16) _Float16 sA[8][16 * LSTR];
  __shared__ __align__(16) _Float16 sB[8][16 * LSTR];
  __shared__ __align__(16) float    sC[8][16 * DPA];
  const int wave = threadIdx.x >> 5, lane = threadIdx.x & 31;
  _Float16* bufA = sA[wave];
  _Float16* bufB = sB[wave];
  float*    bufC = sC[wave];
  const int m = lane & 15;
  const long node = ((long)blockIdx.x * 8 + wave) * 16 + m;

  if (lane < 16) {
    const float* xp = x + node * DL;
#pragma unroll
    for (int i = 0; i < 16; ++i) bufA[m * LSTR + i] = (_Float16)xp[i];
  } else {
#pragma unroll
    for (int i = 0; i < 16; ++i) bufA[m * LSTR + 16 + i] = (_Float16)0.f;
  }
  __syncthreads();
  layer2t(bufA, bufB, pk + 0, b1, DH, lane);
  __syncthreads();
  layer2t(bufB, bufA, pk + 1024, b2, DH, lane);
  __syncthreads();
  layer_final(bufA, bufC, pk + 2048, b3, 1, 2, lane);
  __syncthreads();
  if (lane < 16) {
    out[node * 2 + 0] = bufC[m * DPA + 0];
    out[node * 2 + 1] = bufC[m * DPA + 1];
  }
}

// out_global: u[B,16] -> MLP(16,20,20,4) -> out[B,4]
__global__ void outglobal_kernel(const float* __restrict__ u, float* __restrict__ out,
                                 const float* W1, const float* b1,
                                 const float* W2, const float* b2,
                                 const float* W3, const float* b3, int B) {
  int g = blockIdx.x * blockDim.x + threadIdx.x;
  if (g >= B) return;
  float h0[16];
  for (int i = 0; i < 16; ++i) h0[i] = u[(long)g * 16 + i];
  float h1[20];
  for (int j = 0; j < 20; ++j) {
    float s = b1[j];
    for (int k = 0; k < 16; ++k) s = fmaf(h0[k], W1[k * 20 + j], s);
    h1[j] = fmaxf(s, 0.f);
  }
  float h2[20];
  for (int j = 0; j < 20; ++j) {
    float s = b2[j];
    for (int k = 0; k < 20; ++k) s = fmaf(h1[k], W2[k * 20 + j], s);
    h2[j] = fmaxf(s, 0.f);
  }
  for (int j = 0; j < 4; ++j) {
    float s = b3[j];
    for (int k = 0; k < 20; ++k) s = fmaf(h2[k], W3[k * 4 + j], s);
    out[(long)g * 4 + j] = s;
  }
}

// ---------------------------------------------------------------------------
// Host side
// ---------------------------------------------------------------------------
extern "C" void kernel_launch(void* const* d_in, const int* in_sizes, int n_in,
                              void* d_out, int out_size, void* d_ws, size_t ws_size,
                              hipStream_t stream) {
  const int B = 10000;
  // Layout detection: insertion-order flatten has batch (2M ints) at idx 1;
  // jax sorted-key flatten has batch_size (scalar) at idx 1.
  const bool ins = (n_in >= 2) && (in_sizes[1] > 1);
  auto F = [&](int a, int b) { return (const float*)d_in[ins ? a : b]; };

  const float* x0  = F(0, 46);
  const int* batch = (const int*)d_in[ins ? 1 : 0];
  const long N = (long)in_sizes[ins ? 0 : 46] / 8;

  const float *iW1 = F(3, 2),  *ib1 = F(4, 5),  *iW2 = F(5, 3),  *ib2 = F(6, 6);
  const float *iW3 = F(7, 4),  *ib3 = F(8, 7),  *ig  = F(9, 9),  *ibe = F(10, 8);
  const float *nW1 = F(11,10), *nb1 = F(12,13), *nW2 = F(13,11), *nb2 = F(14,14);
  const float *nW3 = F(15,12), *nb3 = F(16,15), *ng  = F(17,17), *nbe = F(18,16);
  const float *pW1 = F(19,18), *pb1 = F(20,21), *pW2 = F(21,19), *pb2 = F(22,22);
  const float *pW3 = F(23,20), *pb3 = F(24,23), *pg  = F(25,25), *pbe = F(26,24);
  const float *qW1 = F(27,26), *qb1 = F(28,29), *qW2 = F(29,27), *qb2 = F(30,30);
  const float *qW3 = F(31,28), *qb3 = F(32,31), *qg  = F(33,33), *qbe = F(34,32);
  const float *oW1 = F(35,40), *ob1 = F(36,43), *oW2 = F(37,41), *ob2 = F(38,44);
  const float *oW3 = F(39,42), *ob3 = F(40,45);
  const float *gW1 = F(41,34), *gb1 = F(42,37), *gW2 = F(43,35), *gb2 = F(44,38);
  const float *gW3 = F(45,36), *gb3 = F(46,39);

  // Workspace layout
  char* ws = (char*)d_ws;
  float* xbuf   = (float*)ws;                                    // N*16 f32
  float* ubuf   = (float*)(ws + (size_t)N * 16 * 4);             // B*16 f32
  float* aggbuf = (float*)(ws + (size_t)N * 16 * 4 + (size_t)B * 16 * 4);  // B*20 f32
  size_t pkOff = (((size_t)N * 16 * 4 + (size_t)B * 16 * 4 + (size_t)B * 20 * 4) + 255) &
                 ~(size_t)255;
  _Float16* pkInit = (_Float16*)(ws + pkOff);       // 5 tiles  (2560 halfs)
  _Float16* pkMeta = pkInit + 2560;                 // 20 layers x 11 tiles (5632 halfs each)
  _Float16* pkOut  = pkMeta + (size_t)20 * 5632;    // 5 tiles

  auto pack = [&](const float* W, _Float16* dst, int K, int Nn, int tiles, int layers,
                  long wS, long dS) {
    long total = (long)layers * tiles * 512;
    pack_kernel<<<(int)((total + 255) / 256), 256, 0, stream>>>(W, dst, K, Nn, tiles,
                                                                layers, wS, dS);
  };
  pack(iW1, pkInit + 0,    8,  20, 2, 1, 0, 0);
  pack(iW2, pkInit + 1024, 20, 20, 2, 1, 0, 0);
  pack(iW3, pkInit + 2048, 20, 16, 1, 1, 0, 0);
  pack(nW1, pkMeta + 0,    32, 20, 2, 20, 640, 5632);
  pack(nW2, pkMeta + 1024, 20, 20, 2, 20, 400, 5632);
  pack(nW3, pkMeta + 2048, 20, 16, 1, 20, 320, 5632);
  pack(pW1, pkMeta + 2560, 32, 20, 2, 20, 640, 5632);
  pack(pW2, pkMeta + 3584, 20, 20, 2, 20, 400, 5632);
  pack(pW3, pkMeta + 4608, 20, 20, 2, 20, 400, 5632);
  pack(oW1, pkOut + 0,     16, 20, 2, 1, 0, 0);
  pack(oW2, pkOut + 1024,  20, 20, 2, 1, 0, 0);
  pack(oW3, pkOut + 2048,  20, 2,  1, 1, 0, 0);

  zero2_kernel<<<(B * 20 + 255) / 256, 256, 0, stream>>>(ubuf, (long)B * 16, aggbuf,
                                                         (long)B * 20);

  const int tileBlocks = (int)(N / 128);  // 16 nodes/wave, 8 waves/block
  init_kernel<<<tileBlocks, 256, 0, stream>>>(x0, xbuf, pkInit, ib1, ib2, ib3, ig, ibe);

  for (int l = 0; l < 20; ++l) {
    node_kernel<<<tileBlocks, 256, 0, stream>>>(
        xbuf, batch, ubuf, aggbuf, pkMeta + (long)l * 5632,
        nb1 + l * 20, nb2 + l * 20, nb3 + l * 16, ng + l * 16, nbe + l * 16,
        pb1 + l * 20, pb2 + l * 20, pb3 + l * 20, pg + l * 20, pbe + l * 20);
    global_kernel<<<(B + 255) / 256, 256, 0, stream>>>(
        ubuf, aggbuf, qW1 + l * 720, qb1 + l * 20, qW2 + l * 400, qb2 + l * 20,
        qW3 + l * 320, qb3 + l * 16, qg + l * 16, qbe + l * 16, B);
  }

  float* out = (float*)d_out;
  outlocal_kernel<<<tileBlocks, 256, 0, stream>>>(xbuf, out, pkOut, ob1, ob2, ob3);
  outglobal_kernel<<<(B + 255) / 256, 256, 0, stream>>>(ubuf, out + (size_t)N * 2, gW1,
                                                        gb1, gW2, gb2, gW3, gb3, B);
}
